// EnhancedMultiTokenAggregation_4389456577376
// MI455X (gfx1250) — matrix operand order repaired and checked by reading, not compiled, not run on hardware
//
#include <hip/hip_runtime.h>
#include <cstdint>
#include <cstddef>

typedef __bf16 bf16_t;
typedef __attribute__((ext_vector_type(16))) __bf16 v16bf;
typedef __attribute__((ext_vector_type(8)))  __bf16 v8bf;
typedef __attribute__((ext_vector_type(8)))  float   v8f;

#define HDIM 256
#define BATCH 2048
#define SEQ 200
#define GRP 50        // pooled groups per batch (200/4)
#define POOLN 4
#define NTOK 4
#define HEADD 64
#define PK_RG 8       // groups per pool block
#define PK_RM 32      // rows per pool block (PK_RG * POOLN)
#define LDA 264       // LDS row stride (bf16 elements), multiple of 8

// ---------------- CDNA5 WMMA bf16 16x16x32 fragment helpers ----------------
// A (16x32, row-major in LDS): lane l -> M = l&15 ; halves 0..7 -> K = 8*(l>>4)+e,
// halves 8..15 -> K = 16 + 8*(l>>4) + (e-8)   (per ISA 16-bit A-matrix table)
__device__ __forceinline__ v16bf load_A_frag(const bf16_t* a0, int ld, int lane) {
  int m  = lane & 15;
  int hi = lane >> 4;
  const bf16_t* r = a0 + m * ld + 8 * hi;
  v8bf x0 = *(const v8bf*)(r);
  v8bf x1 = *(const v8bf*)(r + 16);
  v16bf out;
#pragma unroll
  for (int e = 0; e < 8; ++e) { out[e] = x0[e]; out[e + 8] = x1[e]; }
  return out;
}

// B (32x16): lane l -> N = l&15 ; halves 0..7 -> K = 8*(l>>4)+e, halves 8..15 -> K = 16 + 8*(l>>4) + (e-8).
// Weights stored row-major [N_out][K=256] bf16, so each lane does one 32B load pair.
__device__ __forceinline__ v16bf load_B_frag(const bf16_t* W, int ldk, int n0, int k0, int lane) {
  const bf16_t* p = W + (size_t)(n0 + (lane & 15)) * ldk + k0 + 8 * (lane >> 4);
  v8bf x0 = *(const v8bf*)(p);
  v8bf x1 = *(const v8bf*)(p + 16);
  v16bf out;
#pragma unroll
  for (int e = 0; e < 8; ++e) { out[e] = x0[e]; out[e + 8] = x1[e]; }
  return out;
}

// One 16x16 D tile, K=256 (8 WMMA steps). D element r -> row 8*(l>>4)+r, col l&15.
__device__ __forceinline__ v8f wmma_k256(const bf16_t* Atile, int ld,
                                         const bf16_t* W, int n0, int lane) {
  v8f c = {};
#pragma unroll
  for (int k0 = 0; k0 < 256; k0 += 32) {
    v16bf a = load_A_frag(Atile + k0, ld, lane);
    v16bf b = load_B_frag(W, 256, n0, k0, lane);
    c = __builtin_amdgcn_wmma_f32_16x16x32_bf16(false, a, false, b, (short)0, c, false, false);
  }
  return c;
}

// ---------------- weight conversion ----------------
__global__ void convert_bf16_kernel(const float* __restrict__ src, bf16_t* __restrict__ dst, int n) {
  int i = blockIdx.x * 256 + threadIdx.x;
  if (i < n) dst[i] = (bf16_t)src[i];
}

// ---------------- constants: pooled query head + GQA token queries ----------------
__global__ void const_kernel(const float* __restrict__ pq, const float* __restrict__ inw,
                             const float* __restrict__ inb, const float* __restrict__ tok,
                             const float* __restrict__ qw, const float* __restrict__ qb,
                             float* __restrict__ qp, float* __restrict__ q2c) {
  int o = threadIdx.x;              // 256 threads
  float a = inb[o];
  for (int k = 0; k < 256; ++k) a += inw[o * 256 + k] * pq[k];
  qp[o] = a;
  for (int t = 0; t < NTOK; ++t) {
    float c = qb[o];
    for (int k = 0; k < 256; ++k) c += qw[o * 256 + k] * tok[t * 256 + k];
    q2c[t * 256 + o] = c;
  }
}

// ---------------- fused sequence pooling: X -> pooled[B*GRP, 256] ----------------
__global__ __launch_bounds__(256) void pool_kernel(
    const float* __restrict__ X, const bf16_t* __restrict__ Wkv,
    const float* __restrict__ bkv, const float* __restrict__ qp,
    const bf16_t* __restrict__ Wo, const float* __restrict__ bo,
    float* __restrict__ pooled)
{
  __shared__ bf16_t xbf[PK_RM][LDA];
  __shared__ float  kv[PK_RM][512];     // K cols 0..255, V cols 256..511
  __shared__ bf16_t obf[16][LDA];       // 8 group outputs, padded to 16 rows

  const int tid  = threadIdx.x;
  const int lane = tid & 31;
  const int wave = tid >> 5;
  const int g0   = blockIdx.x * PK_RG;
  const int row0 = g0 * POOLN;

  // load & convert X tile (32 rows x 256)
  for (int i = tid; i < PK_RM * 64; i += 256) {
    int r = i >> 6, c4 = i & 63;
    float4 v = ((const float4*)(X + (size_t)(row0 + r) * HDIM))[c4];
    bf16_t* d = &xbf[r][c4 * 4];
    d[0] = (bf16_t)v.x; d[1] = (bf16_t)v.y; d[2] = (bf16_t)v.z; d[3] = (bf16_t)v.w;
  }
  for (int i = tid; i < 8 * LDA; i += 256) (&obf[8][0])[i] = (bf16_t)0.f;
  __syncthreads();

  // KV GEMM: 2 M-tiles x 32 N-tiles
  for (int t = wave; t < 64; t += 8) {
    int mt = t >> 5, nt = t & 31;
    v8f c = wmma_k256(&xbf[mt * 16][0], LDA, Wkv, nt * 16, lane);
    int n = nt * 16 + (lane & 15);
    float bias = bkv[n];
    int mbase = mt * 16 + 8 * (lane >> 4);
#pragma unroll
    for (int r = 0; r < 8; ++r) kv[mbase + r][n] = c[r] + bias;
  }
  __syncthreads();

  // per-group single-query 4-head attention (POOL=4 keys)
  {
    int g  = tid >> 5;            // group 0..7 (one wave per group)
    int h  = (tid >> 3) & 3;      // head
    int qd = tid & 7;             // d-octant (8 dims each)
    int d0 = h * HEADD + qd * 8;
    float s0 = 0.f, s1 = 0.f, s2 = 0.f, s3 = 0.f;
#pragma unroll
    for (int d = 0; d < 8; ++d) {
      float q = qp[d0 + d];
      s0 += q * kv[g * 4 + 0][d0 + d];
      s1 += q * kv[g * 4 + 1][d0 + d];
      s2 += q * kv[g * 4 + 2][d0 + d];
      s3 += q * kv[g * 4 + 3][d0 + d];
    }
#pragma unroll
    for (int off = 1; off <= 4; off <<= 1) {
      s0 += __shfl_xor(s0, off, 32);
      s1 += __shfl_xor(s1, off, 32);
      s2 += __shfl_xor(s2, off, 32);
      s3 += __shfl_xor(s3, off, 32);
    }
    s0 *= 0.125f; s1 *= 0.125f; s2 *= 0.125f; s3 *= 0.125f;   // 1/sqrt(64)
    float m = fmaxf(fmaxf(s0, s1), fmaxf(s2, s3));
    float e0 = __expf(s0 - m), e1 = __expf(s1 - m), e2 = __expf(s2 - m), e3 = __expf(s3 - m);
    float inv = 1.f / (e0 + e1 + e2 + e3);
#pragma unroll
    for (int d = 0; d < 8; ++d) {
      int c = 256 + d0 + d;
      float o = (e0 * kv[g * 4 + 0][c] + e1 * kv[g * 4 + 1][c] +
                 e2 * kv[g * 4 + 2][c] + e3 * kv[g * 4 + 3][c]) * inv;
      obf[g][d0 + d] = (bf16_t)o;
    }
  }
  __syncthreads();

  // output projection [8(pad16),256] x [256,256]
  for (int t = wave; t < 16; t += 8) {
    v8f c = wmma_k256(&obf[0][0], LDA, Wo, t * 16, lane);
    int n = t * 16 + (lane & 15), hi = lane >> 4;
    float bias = bo[n];
#pragma unroll
    for (int r = 0; r < 8; ++r) {
      int row = 8 * hi + r;
      if (row < PK_RG)
        pooled[(size_t)(g0 + row) * HDIM + n] = c[r] + bias;
    }
  }
}

// ---------------- GQA over pooled sequence (block per batch) ----------------
__global__ __launch_bounds__(256) void gqa_kernel(
    const float* __restrict__ pooled, const bf16_t* __restrict__ Wkv2,
    const float* __restrict__ kb, const float* __restrict__ vb,
    const float* __restrict__ q2c, const bf16_t* __restrict__ Wo,
    const float* __restrict__ bo, float* __restrict__ out)
{
  __shared__ bf16_t pbf[64][LDA];       // 50 rows padded to 64
  __shared__ float  kv2[64][128];       // K cols 0..63, V cols 64..127
  __shared__ float  wts[16][52];
  __shared__ bf16_t abf[16][LDA];       // 4 token rows padded to 16

  const int tid = threadIdx.x, lane = tid & 31, wave = tid >> 5;
  const int b = blockIdx.x;
  const float* P = pooled + (size_t)b * GRP * HDIM;

  for (int i = tid; i < 64 * 64; i += 256) {
    int r = i >> 6, c4 = i & 63;
    bf16_t* d = &pbf[r][c4 * 4];
    if (r < GRP) {
      float4 v = ((const float4*)(P + (size_t)r * HDIM))[c4];
      d[0] = (bf16_t)v.x; d[1] = (bf16_t)v.y; d[2] = (bf16_t)v.z; d[3] = (bf16_t)v.w;
    } else {
      d[0] = d[1] = d[2] = d[3] = (bf16_t)0.f;
    }
  }
  for (int i = tid; i < 16 * LDA; i += 256) (&abf[0][0])[i] = (bf16_t)0.f;
  __syncthreads();

  // KV GEMM: 4 M-tiles x 8 N-tiles
  for (int t = wave; t < 32; t += 8) {
    int mt = t >> 3, nt = t & 7;
    v8f c = wmma_k256(&pbf[mt * 16][0], LDA, Wkv2, nt * 16, lane);
    int n = nt * 16 + (lane & 15);
    float bias = (n < 64) ? kb[n] : vb[n - 64];
    int mbase = mt * 16 + 8 * (lane >> 4);
#pragma unroll
    for (int r = 0; r < 8; ++r) kv2[mbase + r][n] = c[r] + bias;
  }
  __syncthreads();

  // single-KV-head attention: 16 (token,head) pairs, 16 lanes each
  {
    int p = tid >> 4, sub = tid & 15;
    int tt = p >> 2, h = p & 3;
    const float* q = q2c + tt * HDIM + h * HEADD;
    float sc[4];
#pragma unroll
    for (int jj = 0; jj < 4; ++jj) {
      int s = sub + jj * 16;
      float a = -1e30f;
      if (s < GRP) {
        a = 0.f;
        for (int d = 0; d < HEADD; ++d) a += q[d] * kv2[s][d];
        a *= 0.125f;                                   // 1/sqrt(64)
      }
      sc[jj] = a;
    }
    float m = fmaxf(fmaxf(sc[0], sc[1]), fmaxf(sc[2], sc[3]));
#pragma unroll
    for (int off = 1; off <= 8; off <<= 1) m = fmaxf(m, __shfl_xor(m, off, 32));
    float ee[4]; float lsum = 0.f;
#pragma unroll
    for (int jj = 0; jj < 4; ++jj) {
      int s = sub + jj * 16;
      ee[jj] = (s < GRP) ? __expf(sc[jj] - m) : 0.f;
      lsum += ee[jj];
    }
#pragma unroll
    for (int off = 1; off <= 8; off <<= 1) lsum += __shfl_xor(lsum, off, 32);
    float inv = 1.f / lsum;
#pragma unroll
    for (int jj = 0; jj < 4; ++jj) {
      int s = sub + jj * 16;
      if (s < GRP) wts[p][s] = ee[jj] * inv;
    }
  }
  __syncthreads();
  {
    int p = tid >> 4, sub = tid & 15;
    int tt = p >> 2, h = p & 3;
    float o0 = 0.f, o1 = 0.f, o2 = 0.f, o3 = 0.f;
    for (int s = 0; s < GRP; ++s) {
      float w = wts[p][s];
      const float* v = &kv2[s][64 + sub * 4];
      o0 += w * v[0]; o1 += w * v[1]; o2 += w * v[2]; o3 += w * v[3];
    }
    bf16_t* d = &abf[tt][h * HEADD + sub * 4];
    d[0] = (bf16_t)o0; d[1] = (bf16_t)o1; d[2] = (bf16_t)o2; d[3] = (bf16_t)o3;
  }
  __syncthreads();

  // output projection
  for (int t = wave; t < 16; t += 8) {
    v8f c = wmma_k256(&abf[0][0], LDA, Wo, t * 16, lane);
    int n = t * 16 + (lane & 15), hi = lane >> 4;
    float bias = bo[n];
#pragma unroll
    for (int r = 0; r < 8; ++r) {
      int row = 8 * hi + r;
      if (row < NTOK) out[((size_t)b * NTOK + row) * HDIM + n] = c[r] + bias;
    }
  }
}

// ---------------- combine + topk(=full) attention + LN + out proj ----------------
__global__ __launch_bounds__(256) void final_kernel(
    const float* __restrict__ ua, const float* __restrict__ ia,
    const bf16_t* __restrict__ Wqkv, const float* __restrict__ bq,
    const float* __restrict__ bk, const float* __restrict__ bv,
    const bf16_t* __restrict__ Wio, const float* __restrict__ bio,
    const float* __restrict__ lng, const float* __restrict__ lnb,
    const bf16_t* __restrict__ Wout, const float* __restrict__ bout,
    float* __restrict__ out)
{
  __shared__ bf16_t cbf[16][LDA];
  __shared__ float  comb[4][256];
  __shared__ float  qkvf[4][768];   // q 0..255, k 256..511, v 512..767
  __shared__ bf16_t o3bf[16][LDA];
  __shared__ float  pre[4][256];
  __shared__ bf16_t nbf[16][LDA];

  const int tid = threadIdx.x, lane = tid & 31, wave = tid >> 5;
  const int b = blockIdx.x;

  for (int i = tid; i < 1024; i += 256) {
    float v = ua[(size_t)b * 1024 + i] + ia[(size_t)b * 1024 + i];
    int r = i >> 8, c = i & 255;
    comb[r][c] = v;
    cbf[r][c] = (bf16_t)v;
  }
  for (int i = tid; i < 12 * LDA; i += 256) {
    (&cbf[4][0])[i]  = (bf16_t)0.f;
    (&o3bf[4][0])[i] = (bf16_t)0.f;
    (&nbf[4][0])[i]  = (bf16_t)0.f;
  }
  __syncthreads();

  // QKV GEMM: 48 N-tiles
  for (int t = wave; t < 48; t += 8) {
    v8f c = wmma_k256(&cbf[0][0], LDA, Wqkv, t * 16, lane);
    int n = t * 16 + (lane & 15), hi = lane >> 4;
    float bias = (n < 256) ? bq[n] : (n < 512 ? bk[n - 256] : bv[n - 512]);
#pragma unroll
    for (int r = 0; r < 8; ++r) {
      int row = 8 * hi + r;
      if (row < 4) qkvf[row][n] = c[r] + bias;
    }
  }
  __syncthreads();

  // 4x4 full softmax attention (top-k with k == T is a permutation -> identical)
  {
    int p = tid >> 4, sub = tid & 15;
    int tt = p >> 2, h = p & 3;
    int s = sub & 3, dq = sub >> 2;
    const float* q  = &qkvf[tt][h * HEADD + dq * 16];
    const float* kk = &qkvf[s][256 + h * HEADD + dq * 16];
    float sc = 0.f;
#pragma unroll
    for (int d = 0; d < 16; ++d) sc += q[d] * kk[d];
    sc += __shfl_xor(sc, 4, 32);
    sc += __shfl_xor(sc, 8, 32);
    sc *= 0.125f;
    float m = sc;
    m = fmaxf(m, __shfl_xor(m, 1, 32));
    m = fmaxf(m, __shfl_xor(m, 2, 32));
    float e = __expf(sc - m);
    float sum = e;
    sum += __shfl_xor(sum, 1, 32);
    sum += __shfl_xor(sum, 2, 32);
    float w = e / sum;
    const float* vr = &qkvf[s][512 + h * HEADD + dq * 16];
    float ov[16];
#pragma unroll
    for (int d = 0; d < 16; ++d) ov[d] = w * vr[d];
#pragma unroll
    for (int d = 0; d < 16; ++d) {
      ov[d] += __shfl_xor(ov[d], 1, 32);
      ov[d] += __shfl_xor(ov[d], 2, 32);
    }
    if ((sub & 3) == 0) {
      bf16_t* dptr = &o3bf[tt][h * HEADD + dq * 16];
#pragma unroll
      for (int d = 0; d < 16; ++d) dptr[d] = (bf16_t)ov[d];
    }
  }
  __syncthreads();

  // inter projection + residual
  for (int t = wave; t < 16; t += 8) {
    v8f c = wmma_k256(&o3bf[0][0], LDA, Wio, t * 16, lane);
    int n = t * 16 + (lane & 15), hi = lane >> 4;
    float bias = bio[n];
#pragma unroll
    for (int r = 0; r < 8; ++r) {
      int row = 8 * hi + r;
      if (row < 4) pre[row][n] = c[r] + bias + comb[row][n];
    }
  }
  __syncthreads();

  // LayerNorm: one wave per token row
  if (wave < 4) {
    int row = wave;
    float x[8]; float sum = 0.f;
#pragma unroll
    for (int k = 0; k < 8; ++k) { x[k] = pre[row][lane + 32 * k]; sum += x[k]; }
#pragma unroll
    for (int off = 1; off <= 16; off <<= 1) sum += __shfl_xor(sum, off, 32);
    float mu = sum * (1.f / 256.f);
    float vs = 0.f;
#pragma unroll
    for (int k = 0; k < 8; ++k) { float d = x[k] - mu; vs += d * d; }
#pragma unroll
    for (int off = 1; off <= 16; off <<= 1) vs += __shfl_xor(vs, off, 32);
    float rs = rsqrtf(vs * (1.f / 256.f) + 1e-5f);
#pragma unroll
    for (int k = 0; k < 8; ++k) {
      int c = lane + 32 * k;
      nbf[row][c] = (bf16_t)((x[k] - mu) * rs * lng[c] + lnb[c]);
    }
  }
  __syncthreads();

  // final projection -> out
  for (int t = wave; t < 16; t += 8) {
    v8f c = wmma_k256(&nbf[0][0], LDA, Wout, t * 16, lane);
    int n = t * 16 + (lane & 15), hi = lane >> 4;
    float bias = bout[n];
#pragma unroll
    for (int r = 0; r < 8; ++r) {
      int row = 8 * hi + r;
      if (row < 4) out[((size_t)b * 4 + row) * HDIM + n] = c[r] + bias;
    }
  }
}

// ---------------- host launch ----------------
extern "C" void kernel_launch(void* const* d_in, const int* in_sizes, int n_in,
                              void* d_out, int out_size, void* d_ws, size_t ws_size,
                              hipStream_t stream) {
  (void)in_sizes; (void)n_in; (void)out_size; (void)ws_size;
  const float* user_seq = (const float*)d_in[0];
  const float* item_seq = (const float*)d_in[1];
  const float* agg      = (const float*)d_in[2];
  const float* up_query = (const float*)d_in[3];
  const float* up_inw   = (const float*)d_in[4];
  const float* up_inb   = (const float*)d_in[5];
  const float* up_ow    = (const float*)d_in[6];
  const float* up_ob    = (const float*)d_in[7];
  const float* ip_query = (const float*)d_in[8];
  const float* ip_inw   = (const float*)d_in[9];
  const float* ip_inb   = (const float*)d_in[10];
  const float* ip_ow    = (const float*)d_in[11];
  const float* ip_ob    = (const float*)d_in[12];
  const float* ug_qw = (const float*)d_in[13]; const float* ug_qb = (const float*)d_in[14];
  const float* ug_kw = (const float*)d_in[15]; const float* ug_kb = (const float*)d_in[16];
  const float* ug_vw = (const float*)d_in[17]; const float* ug_vb = (const float*)d_in[18];
  const float* ug_ow = (const float*)d_in[19]; const float* ug_ob = (const float*)d_in[20];
  const float* ig_qw = (const float*)d_in[21]; const float* ig_qb = (const float*)d_in[22];
  const float* ig_kw = (const float*)d_in[23]; const float* ig_kb = (const float*)d_in[24];
  const float* ig_vw = (const float*)d_in[25]; const float* ig_vb = (const float*)d_in[26];
  const float* ig_ow = (const float*)d_in[27]; const float* ig_ob = (const float*)d_in[28];
  const float* ti_qw = (const float*)d_in[29]; const float* ti_qb = (const float*)d_in[30];
  const float* ti_kw = (const float*)d_in[31]; const float* ti_kb = (const float*)d_in[32];
  const float* ti_vw = (const float*)d_in[33]; const float* ti_vb = (const float*)d_in[34];
  const float* ti_ow = (const float*)d_in[35]; const float* ti_ob = (const float*)d_in[36];
  const float* ln_g  = (const float*)d_in[37]; const float* ln_b  = (const float*)d_in[38];
  const float* out_w = (const float*)d_in[39]; const float* out_b = (const float*)d_in[40];

  char* ws = (char*)d_ws;
  size_t off = 0;
  auto take = [&](size_t bytes) -> void* {
    void* p = ws + off;
    off += (bytes + 255) & ~(size_t)255;
    return p;
  };

  bf16_t* wkv_u = (bf16_t*)take(512 * 256 * sizeof(bf16_t));
  bf16_t* wkv_i = (bf16_t*)take(512 * 256 * sizeof(bf16_t));
  bf16_t* pow_u = (bf16_t*)take(256 * 256 * sizeof(bf16_t));
  bf16_t* pow_i = (bf16_t*)take(256 * 256 * sizeof(bf16_t));
  bf16_t* gkv_u = (bf16_t*)take(128 * 256 * sizeof(bf16_t));
  bf16_t* gkv_i = (bf16_t*)take(128 * 256 * sizeof(bf16_t));
  bf16_t* gow_u = (bf16_t*)take(256 * 256 * sizeof(bf16_t));
  bf16_t* gow_i = (bf16_t*)take(256 * 256 * sizeof(bf16_t));
  bf16_t* tqkv  = (bf16_t*)take(768 * 256 * sizeof(bf16_t));
  bf16_t* tow   = (bf16_t*)take(256 * 256 * sizeof(bf16_t));
  bf16_t* outw  = (bf16_t*)take(256 * 256 * sizeof(bf16_t));
  float* qp_u  = (float*)take(256 * sizeof(float));
  float* qp_i  = (float*)take(256 * sizeof(float));
  float* q2c_u = (float*)take(4 * 256 * sizeof(float));
  float* q2c_i = (float*)take(4 * 256 * sizeof(float));
  float* pooled_u = (float*)take((size_t)BATCH * GRP * HDIM * sizeof(float));
  float* pooled_i = (float*)take((size_t)BATCH * GRP * HDIM * sizeof(float));
  float* ua = (float*)take((size_t)BATCH * NTOK * HDIM * sizeof(float));
  float* ia = (float*)take((size_t)BATCH * NTOK * HDIM * sizeof(float));

  auto conv = [&](const float* s, bf16_t* d, int n) {
    convert_bf16_kernel<<<(n + 255) / 256, 256, 0, stream>>>(s, d, n);
  };
  conv(up_inw + 256 * 256, wkv_u, 512 * 256);   // pool Wk;Wv (rows 256..767)
  conv(ip_inw + 256 * 256, wkv_i, 512 * 256);
  conv(up_ow, pow_u, 256 * 256);
  conv(ip_ow, pow_i, 256 * 256);
  conv(ug_kw, gkv_u, 64 * 256);
  conv(ug_vw, gkv_u + 64 * 256, 64 * 256);
  conv(ig_kw, gkv_i, 64 * 256);
  conv(ig_vw, gkv_i + 64 * 256, 64 * 256);
  conv(ug_ow, gow_u, 256 * 256);
  conv(ig_ow, gow_i, 256 * 256);
  conv(ti_qw, tqkv, 256 * 256);
  conv(ti_kw, tqkv + 256 * 256, 256 * 256);
  conv(ti_vw, tqkv + 512 * 256, 256 * 256);
  conv(ti_ow, tow, 256 * 256);
  conv(out_w, outw, 256 * 256);

  const_kernel<<<1, 256, 0, stream>>>(up_query, up_inw, up_inb, agg, ug_qw, ug_qb, qp_u, q2c_u);
  const_kernel<<<1, 256, 0, stream>>>(ip_query, ip_inw, ip_inb, agg, ig_qw, ig_qb, qp_i, q2c_i);

  pool_kernel<<<(BATCH * GRP) / PK_RG, 256, 0, stream>>>(
      user_seq, wkv_u, up_inb + 256, qp_u, pow_u, up_ob, pooled_u);
  pool_kernel<<<(BATCH * GRP) / PK_RG, 256, 0, stream>>>(
      item_seq, wkv_i, ip_inb + 256, qp_i, pow_i, ip_ob, pooled_i);

  gqa_kernel<<<BATCH, 256, 0, stream>>>(pooled_u, gkv_u, ug_kb, ug_vb, q2c_u, gow_u, ug_ob, ua);
  gqa_kernel<<<BATCH, 256, 0, stream>>>(pooled_i, gkv_i, ig_kb, ig_vb, q2c_i, gow_i, ig_ob, ia);

  final_kernel<<<BATCH, 256, 0, stream>>>(ua, ia, tqkv, ti_qb, ti_kb, ti_vb,
                                          tow, ti_ob, ln_g, ln_b, outw, out_b,
                                          (float*)d_out);
}
